// Mamba2Simple_15745350107165
// MI455X (gfx1250) — matrix operand order repaired and checked by reading, not compiled, not run on hardware
//
#include <hip/hip_runtime.h>
#include <math.h>

// ---------------- problem constants ----------------
#define ROWS   8192      // B*L = 2*4096
#define DMODEL 512
#define DINNER 1024
#define NHEADS 64
#define DSTATE 16
#define CONVD  1056      // D_INNER + 2*NGROUPS*D_STATE
#define DPROJ  2144      // 2*D_INNER + 2*NGROUPS*D_STATE + NHEADS
#define CHUNK  256
#define NC     16        // chunks per batch (4096/256)
#define NBC    32        // B * NC

typedef __attribute__((ext_vector_type(16))) __bf16 bf16x16;
typedef __attribute__((ext_vector_type(8)))  float  f32x8;

union Frag { bf16x16 v; unsigned short u[16]; };

__device__ __forceinline__ unsigned short f2bf(float f) {
  unsigned int x = __builtin_bit_cast(unsigned int, f);
  x += 0x7FFFu + ((x >> 16) & 1u);          // round-to-nearest-even
  return (unsigned short)(x >> 16);
}

__device__ __forceinline__ f32x8 wmma_bf16(Frag a, Frag b, f32x8 c) {
  return __builtin_amdgcn_wmma_f32_16x16x32_bf16(false, a.v, false, b.v,
                                                 (short)0, c, false, false);
}

// =====================================================================
// Kernel: C[M,N] = A[M,K] @ W[N,K]^T, fp32 in/out, bf16 WMMA compute.
// Block: 256 threads (8 waves, 2x4 wave grid). Block tile M=128, N=128,
// K-step 32. Wave tile 64x32 -> 8 accumulators, 8 WMMAs per K-step.
// float4 global loads + global_prefetch of next K-slice.
// =====================================================================
__global__ __launch_bounds__(256)
void gemm_bf16_wmma(const float* __restrict__ A, const float* __restrict__ W,
                    float* __restrict__ C, int M, int N, int K) {
  __shared__ unsigned short As[128][34];   // [m][k], padded vs bank conflicts
  __shared__ unsigned short Ws[128][34];   // [n][k]
  const int tid  = threadIdx.x;
  const int wave = tid >> 5;
  const int lane = tid & 31;
  const int l    = lane & 15;
  const int hi   = lane >> 4;
  const int m0   = blockIdx.x * 128;
  const int n0   = blockIdx.y * 128;
  const int wm   = wave >> 2;              // 0..1  (M direction, 64 rows)
  const int wn   = wave & 3;               // 0..3  (N direction, 32 cols)

  f32x8 acc[4][2] = {};

  for (int k0 = 0; k0 < K; k0 += 32) {
    // ---- fill LDS: A 128x32, W 128x32, as float4 (global_load_b128)
    for (int i = tid; i < 128 * 8; i += 256) {
      int r = i >> 3, c4 = (i & 7) * 4;
      const float* p = &A[(size_t)(m0 + r) * K + k0 + c4];
      float4 v = *(const float4*)p;
      As[r][c4 + 0] = f2bf(v.x); As[r][c4 + 1] = f2bf(v.y);
      As[r][c4 + 2] = f2bf(v.z); As[r][c4 + 3] = f2bf(v.w);
      if (k0 + 32 < K) __builtin_prefetch(p + 32, 0, 0);   // global_prefetch_b8
    }
    for (int i = tid; i < 128 * 8; i += 256) {
      int r = i >> 3, c4 = (i & 7) * 4;
      int n = n0 + r;
      float4 v = {0.f, 0.f, 0.f, 0.f};
      if (n < N) {
        const float* p = &W[(size_t)n * K + k0 + c4];
        v = *(const float4*)p;
        if (k0 + 32 < K) __builtin_prefetch(p + 32, 0, 0);
      }
      Ws[r][c4 + 0] = f2bf(v.x); Ws[r][c4 + 1] = f2bf(v.y);
      Ws[r][c4 + 2] = f2bf(v.z); Ws[r][c4 + 3] = f2bf(v.w);
    }
    __syncthreads();

    Frag a[4], b[2];
#pragma unroll
    for (int i = 0; i < 4; ++i) {                 // A frags: lane=row, k packed
      int row = wm * 64 + i * 16 + l;
#pragma unroll
      for (int j = 0; j < 8; ++j) {
        a[i].u[j]     = As[row][hi * 16 + j];
        a[i].u[8 + j] = As[row][hi * 16 + 8 + j];
      }
    }
#pragma unroll
    for (int i = 0; i < 2; ++i) {                 // B frags: lane=col, k=hi*16+j
      int nrow = wn * 32 + i * 16 + l;
#pragma unroll
      for (int j = 0; j < 16; ++j) b[i].u[j] = Ws[nrow][hi * 16 + j];
    }
#pragma unroll
    for (int i = 0; i < 4; ++i)
#pragma unroll
      for (int jj = 0; jj < 2; ++jj)
        acc[i][jj] = wmma_bf16(a[i], b[jj], acc[i][jj]);
    __syncthreads();
  }

#pragma unroll
  for (int i = 0; i < 4; ++i) {
    const int rbase = m0 + wm * 64 + i * 16 + hi * 8;   // D: VGPR r -> row hi*8+r
#pragma unroll
    for (int jj = 0; jj < 2; ++jj) {
      int n = n0 + wn * 32 + jj * 16 + l;
      if (n < N) {
#pragma unroll
        for (int r = 0; r < 8; ++r)
          C[(size_t)(rbase + r) * N + n] = acc[i][jj][r];
      }
    }
  }
}

// =====================================================================
// Kernel: depthwise causal conv (K=4) + SiLU on xBC; softplus(dt)+dA.
// One block per row (b,l).
// =====================================================================
__global__ __launch_bounds__(256)
void conv_silu_dt(const float* __restrict__ zx, const float* __restrict__ conv_w,
                  const float* __restrict__ conv_b, const float* __restrict__ dt_bias,
                  const float* __restrict__ A_log,
                  float* __restrict__ xbc, float* __restrict__ dtv,
                  float* __restrict__ dAv) {
  const int row  = blockIdx.x;
  const int lpos = row & 4095;
  for (int c = threadIdx.x; c < CONVD; c += 256) {
    float acc = conv_b[c];
#pragma unroll
    for (int k = 0; k < 4; ++k) {
      int lp = lpos - 3 + k;
      if (lp >= 0)
        acc += zx[(size_t)(row - 3 + k) * DPROJ + DINNER + c] * conv_w[c * 4 + k];
    }
    float s = acc / (1.f + __expf(-acc));                    // SiLU
    xbc[(size_t)row * CONVD + c] = s;
  }
  if (threadIdx.x < NHEADS) {
    int h = threadIdx.x;
    float x = zx[(size_t)row * DPROJ + (DINNER + CONVD) + h] + dt_bias[h];
    float dt = (x > 20.f) ? x : log1pf(__expf(x));           // softplus
    float a  = -__expf(A_log[h]);
    dtv[(size_t)row * NHEADS + h] = dt;
    dAv[(size_t)row * NHEADS + h] = dt * a;
  }
}

// =====================================================================
// Kernel: per (batch*chunk, head) SSD chunk:
//   cumsum(dA) in LDS, Y_diag (flash-attn style, 3 WMMAs/step),
//   chunk state S[n][p] (8 WMMAs), chunk dA sum, A_cs to global.
// scores are computed TRANSPOSED so WMMA D layout == next A-frag layout.
// =====================================================================
__global__ __launch_bounds__(256)
void ssd_chunk(const float* __restrict__ xbc, const float* __restrict__ dtv,
               const float* __restrict__ dAv,
               float* __restrict__ y, float* __restrict__ acs_g,
               float* __restrict__ states, float* __restrict__ csum) {
  __shared__ float           sA[CHUNK];
  __shared__ unsigned short  Xs[CHUNK][17];   // x*dt  [s][p] bf16
  __shared__ unsigned short  Bs[CHUNK][17];   // B     [s][n]
  __shared__ unsigned short  Cs[CHUNK][17];   // C     [s][n]
  __shared__ unsigned short  Bd[CHUNK][17];   // B*decay_state [s][n]

  const int bc   = blockIdx.x;       // b*NC + chunk
  const int h    = blockIdx.y;
  const int row0 = bc * CHUNK;       // == b*4096 + chunk*256
  const int s    = threadIdx.x;

  const size_t row = (size_t)(row0 + s);
  const float dt   = dtv[row * NHEADS + h];
  sA[s] = dAv[row * NHEADS + h];
  float bfv[DSTATE];
#pragma unroll
  for (int p = 0; p < 16; ++p)
    Xs[s][p] = f2bf(xbc[row * CONVD + h * 16 + p] * dt);
#pragma unroll
  for (int n = 0; n < 16; ++n) {
    bfv[n]   = xbc[row * CONVD + DINNER + n];
    Bs[s][n] = f2bf(bfv[n]);
    Cs[s][n] = f2bf(xbc[row * CONVD + DINNER + DSTATE + n]);
  }
  __syncthreads();

  // inclusive scan of dA over the chunk (Hillis-Steele)
  for (int off = 1; off < CHUNK; off <<= 1) {
    float t = (s >= off) ? sA[s - off] : 0.f;
    __syncthreads();
    sA[s] += t;
    __syncthreads();
  }
  acs_g[row * NHEADS + h] = sA[s];
  const float total = sA[CHUNK - 1];
  const float dstate = __expf(total - sA[s]);
#pragma unroll
  for (int n = 0; n < 16; ++n) Bd[s][n] = f2bf(bfv[n] * dstate);
  if (s == CHUNK - 1) csum[(size_t)bc * NHEADS + h] = total;
  __syncthreads();

  const int wave = threadIdx.x >> 5;
  const int lane = threadIdx.x & 31;
  const int l    = lane & 15;
  const int hi   = lane >> 4;

  // ---- chunk state: S[n][p] = sum_s Bd[s][n] * X[s][p]  (wave 0)
  if (wave == 0) {
    f32x8 accS = {};
    for (int s0 = 0; s0 < CHUNK; s0 += 32) {
      Frag aB, bX;
#pragma unroll
      for (int j = 0; j < 8; ++j) {
        aB.u[j]     = Bd[s0 + hi * 16 + j][l];
        aB.u[8 + j] = Bd[s0 + hi * 16 + 8 + j][l];
      }
#pragma unroll
      for (int j = 0; j < 16; ++j) bX.u[j] = Xs[s0 + hi * 16 + j][l];
      accS = wmma_bf16(aB, bX, accS);
    }
#pragma unroll
    for (int r = 0; r < 8; ++r)
      states[((size_t)bc * NHEADS + h) * 256 + (hi * 8 + r) * 16 + l] = accS[r];
  }

  // ---- Y_diag: each wave handles two 16-row tiles
  for (int t = 0; t < 2; ++t) {
    const int l0    = (wave + 8 * t) * 16;
    const float acl = sA[l0 + l];               // A_cs at this lane's output row
    f32x8 accy = {};
    for (int s0 = 0; s0 <= l0 + 15; s0 += 32) {
      // scores^T tiles: rows s (two 16-row tiles), cols l.  K = n (16, pad 32)
      Frag aS0, aS1, bC;
#pragma unroll
      for (int j = 0; j < 8; ++j) {
        aS0.u[j]     = Bs[s0 + l][hi * 8 + j];
        aS1.u[j]     = Bs[s0 + 16 + l][hi * 8 + j];
        aS0.u[8 + j] = 0; aS1.u[8 + j] = 0;     // K pad
      }
#pragma unroll
      for (int j = 0; j < 16; ++j)
        bC.u[j] = (j < 8) ? Cs[l0 + l][hi * 8 + j] : (unsigned short)0;
      f32x8 z = {};
      f32x8 t0 = wmma_bf16(aS0, bC, z);
      f32x8 t1 = wmma_bf16(aS1, bC, z);

      // causal mask + decay; D layout of scores^T == A-frag layout of T
      Frag aT;
#pragma unroll
      for (int r = 0; r < 8; ++r) {
        int s1 = s0 + hi * 8 + r;
        int s2 = s1 + 16;
        float v1 = (s1 <= l0 + l) ? t0[r] * __expf(acl - sA[s1]) : 0.f;
        float v2 = (s2 <= l0 + l) ? t1[r] * __expf(acl - sA[s2]) : 0.f;
        aT.u[r]     = f2bf(v1);
        aT.u[8 + r] = f2bf(v2);
      }
      Frag bX;
#pragma unroll
      for (int j = 0; j < 8; ++j) {
        bX.u[j]     = Xs[s0 + hi * 8 + j][l];
        bX.u[8 + j] = Xs[s0 + 16 + hi * 8 + j][l];
      }
      accy = wmma_bf16(aT, bX, accy);
    }
#pragma unroll
    for (int r = 0; r < 8; ++r)
      y[(size_t)(row0 + l0 + hi * 8 + r) * DINNER + h * 16 + l] = accy[r];
  }
}

// =====================================================================
// Kernel: sequential inter-chunk state recurrence.
// One block per (b,h); thread e owns state element e = n*16+p.
// =====================================================================
__global__ __launch_bounds__(256)
void chunk_scan(const float* __restrict__ states, const float* __restrict__ csum,
                float* __restrict__ prevs) {
  const int bh = blockIdx.x;           // b*NHEADS + h
  const int b  = bh >> 6, h = bh & 63;
  const int e  = threadIdx.x;
  float run = 0.f;
  for (int c = 0; c < NC; ++c) {
    size_t idx = ((size_t)(b * NC + c) * NHEADS + h) * 256 + e;
    prevs[idx] = run;
    run = run * __expf(csum[(size_t)(b * NC + c) * NHEADS + h]) + states[idx];
  }
}

// =====================================================================
// Kernel: Y_off = (C @ prev_state) * exp(A_cs) + x*D_skip, added into y.
// One block per (batch*chunk, head). One WMMA per 16-row tile (K=16 pad 32).
// =====================================================================
__global__ __launch_bounds__(256)
void ssd_off(const float* __restrict__ xbc, const float* __restrict__ acs_g,
             const float* __restrict__ prevs, const float* __restrict__ D_skip,
             float* __restrict__ y) {
  __shared__ unsigned short Cc[CHUNK][17];
  __shared__ unsigned short Ps[16][17];
  const int bc   = blockIdx.x;
  const int h    = blockIdx.y;
  const int row0 = bc * CHUNK;
  const int s    = threadIdx.x;
#pragma unroll
  for (int n = 0; n < 16; ++n)
    Cc[s][n] = f2bf(xbc[(size_t)(row0 + s) * CONVD + DINNER + DSTATE + n]);
  if (s < 256)
    Ps[s >> 4][s & 15] = f2bf(prevs[((size_t)bc * NHEADS + h) * 256 + s]);
  __syncthreads();

  const int wave = threadIdx.x >> 5;
  const int lane = threadIdx.x & 31;
  const int l    = lane & 15;
  const int hi   = lane >> 4;
  const float dsk = D_skip[h];

  for (int t = 0; t < 2; ++t) {
    const int l0 = (wave + 8 * t) * 16;
    Frag aC, bP;
#pragma unroll
    for (int j = 0; j < 8; ++j) {
      aC.u[j]     = Cc[l0 + l][hi * 8 + j];
      aC.u[8 + j] = 0;
    }
#pragma unroll
    for (int j = 0; j < 16; ++j) bP.u[j] = (j < 8) ? Ps[hi * 8 + j][l] : (unsigned short)0;
    f32x8 z = {};
    f32x8 yo = wmma_bf16(aC, bP, z);
#pragma unroll
    for (int r = 0; r < 8; ++r) {
      size_t row = (size_t)(row0 + l0 + hi * 8 + r);
      float sd = __expf(acs_g[row * NHEADS + h]);
      float xv = xbc[row * CONVD + h * 16 + l];
      y[row * DINNER + h * 16 + l] += yo[r] * sd + xv * dsk;
    }
  }
}

// =====================================================================
// Kernel: y * silu(z), RMSNorm over 1024, scale by norm_w. One block/row.
// =====================================================================
__global__ __launch_bounds__(256)
void gate_norm(const float* __restrict__ y, const float* __restrict__ zx,
               const float* __restrict__ norm_w, float* __restrict__ yn) {
  __shared__ float red[256];
  const int row = blockIdx.x;
  const int tid = threadIdx.x;
  float g[4]; float ss = 0.f;
#pragma unroll
  for (int i = 0; i < 4; ++i) {
    int d = tid + i * 256;
    float zv = zx[(size_t)row * DPROJ + d];
    float gv = y[(size_t)row * DINNER + d] * (zv / (1.f + __expf(-zv)));
    g[i] = gv; ss += gv * gv;
  }
  red[tid] = ss; __syncthreads();
  for (int o = 128; o > 0; o >>= 1) {
    if (tid < o) red[tid] += red[tid + o];
    __syncthreads();
  }
  float r = rsqrtf(red[0] / (float)DINNER + 1e-5f);
#pragma unroll
  for (int i = 0; i < 4; ++i) {
    int d = tid + i * 256;
    yn[(size_t)row * DINNER + d] = g[i] * r * norm_w[d];
  }
}

// =====================================================================
extern "C" void kernel_launch(void* const* d_in, const int* in_sizes, int n_in,
                              void* d_out, int out_size, void* d_ws, size_t ws_size,
                              hipStream_t stream) {
  (void)in_sizes; (void)n_in; (void)out_size; (void)ws_size;
  const float* u          = (const float*)d_in[0];
  const float* in_proj_w  = (const float*)d_in[1];
  const float* conv_w     = (const float*)d_in[2];
  const float* conv_b     = (const float*)d_in[3];
  const float* dt_bias    = (const float*)d_in[4];
  const float* A_log      = (const float*)d_in[5];
  const float* D_skip     = (const float*)d_in[6];
  const float* norm_w     = (const float*)d_in[7];
  const float* out_proj_w = (const float*)d_in[8];

  float* ws  = (float*)d_ws;
  float* zx  = ws;                                  // [8192][2144]
  float* xbc = zx  + (size_t)ROWS * DPROJ;          // [8192][1056]
  float* dtv = xbc + (size_t)ROWS * CONVD;          // [8192][64]
  float* dAv = dtv + (size_t)ROWS * NHEADS;         // [8192][64]
  float* acs = dAv + (size_t)ROWS * NHEADS;         // [8192][64]
  float* yb  = acs + (size_t)ROWS * NHEADS;         // [8192][1024]
  float* st  = yb  + (size_t)ROWS * DINNER;         // [32][64][256]
  float* pv  = st  + (size_t)NBC * NHEADS * 256;    // [32][64][256]
  float* cs  = pv  + (size_t)NBC * NHEADS * 256;    // [32][64]
  float* yn  = cs  + (size_t)NBC * NHEADS;          // [8192][1024]

  dim3 blk(256);
  gemm_bf16_wmma<<<dim3(ROWS / 128, (DPROJ + 127) / 128), blk, 0, stream>>>(
      u, in_proj_w, zx, ROWS, DPROJ, DMODEL);
  conv_silu_dt<<<dim3(ROWS), blk, 0, stream>>>(
      zx, conv_w, conv_b, dt_bias, A_log, xbc, dtv, dAv);
  ssd_chunk<<<dim3(NBC, NHEADS), blk, 0, stream>>>(
      xbc, dtv, dAv, yb, acs, st, cs);
  chunk_scan<<<dim3(2 * NHEADS), blk, 0, stream>>>(st, cs, pv);
  ssd_off<<<dim3(NBC, NHEADS), blk, 0, stream>>>(xbc, acs, pv, D_skip, yb);
  gate_norm<<<dim3(ROWS), blk, 0, stream>>>(yb, zx, norm_w, yn);
  gemm_bf16_wmma<<<dim3(ROWS / 128, DMODEL / 128), blk, 0, stream>>>(
      yn, out_proj_w, (float*)d_out, ROWS, DMODEL, DINNER);
}
